// WindowAttention_1331439862133
// MI455X (gfx1250) — compile-verified, matrix operand order checked by reading, not run on hardware
//
#include <hip/hip_runtime.h>

// ---------------------------------------------------------------------------
// Problem constants (from reference): WS=8, N=64 tokens/window, C=128, NH=4,
// HD=32, H=W=256, B=2 batches -> 2048 windows, 131072 pixels.
// ---------------------------------------------------------------------------
#define IMGDIM   256
#define NWIN     2048
#define IMGELEMS 16777216   // 2*256*256*128 floats per image buffer

typedef __attribute__((ext_vector_type(16))) __bf16          v16bf;
typedef __attribute__((ext_vector_type(8)))  float           v8f;
typedef __attribute__((ext_vector_type(16))) unsigned short  v16u;
typedef __attribute__((ext_vector_type(8)))  unsigned short  v8u;

__device__ __forceinline__ unsigned short f2bf(float f) {
  unsigned u = __float_as_uint(f);
  return (unsigned short)((u + 0x7FFFu + ((u >> 16) & 1u)) >> 16);  // RNE
}
__device__ __forceinline__ float bf2f(unsigned short h) {
  return __uint_as_float(((unsigned)h) << 16);
}

// 16x32 bf16 operand fragment (A layout; also B when the operand is stored
// N-major so fragment "row" == WMMA column).  Per CDNA5 ISA 16-bit layout:
// lanes 0-15 : row = lane,    K halves {0..7, 16..23}
// lanes 16-31: row = lane-16, K halves {8..15, 24..31}
__device__ __forceinline__ v16bf load_frag(const unsigned short* base, int stride, int lane) {
  int row = lane & 15;
  int kb  = (lane >> 4) << 3;            // 0 or 8
  const unsigned short* p = base + row * stride + kb;
  v8u lo = *(const v8u*)(p);             // K kb..kb+7      (16B aligned)
  v8u hi = *(const v8u*)(p + 16);        // K kb+16..kb+23
  v16u t = __builtin_shufflevector(lo, hi, 0,1,2,3,4,5,6,7,8,9,10,11,12,13,14,15);
  return __builtin_bit_cast(v16bf, t);
}

__device__ __forceinline__ v8f wmma_bf16(v16bf a, v16bf b, v8f c) {
  return __builtin_amdgcn_wmma_f32_16x16x32_bf16(false, a, false, b, (short)0, c, false, false);
}

__device__ __forceinline__ v8f vzero8() {
  v8f z;
  #pragma unroll
  for (int i = 0; i < 8; ++i) z[i] = 0.0f;
  return z;
}

// ---------------------------------------------------------------------------
// Weight prep: f32 -> bf16. qkv_w (384,128) and proj_w (128,128) are already
// out-major (einsum 'bhwc,oc->bhwo'). Conv weights HWIO (3,3,128,128) are
// transposed to [tap][cout][cin] so B fragments load N-major.
// ---------------------------------------------------------------------------
__global__ __launch_bounds__(256) void prep_weights(
    const float* __restrict__ qkvw, const float* __restrict__ projw,
    const float* __restrict__ w11, const float* __restrict__ w12,
    const float* __restrict__ w21, const float* __restrict__ w22,
    unsigned short* __restrict__ wq, unsigned short* __restrict__ wp,
    unsigned short* __restrict__ o11, unsigned short* __restrict__ o12,
    unsigned short* __restrict__ o21, unsigned short* __restrict__ o22)
{
  int i = blockIdx.x * 256 + threadIdx.x;
  if (i < 49152) wq[i] = f2bf(qkvw[i]);
  if (i < 16384) wp[i] = f2bf(projw[i]);
  if (i < 147456) {
    int tap = i >> 14;
    int q   = i & 16383;
    int co  = q >> 7;
    int ci  = q & 127;
    int src = (tap << 14) + (ci << 7) + co;   // HWIO source
    o11[i] = f2bf(w11[src]);
    o12[i] = f2bf(w12[src]);
    o21[i] = f2bf(w21[src]);
    o22[i] = f2bf(w22[src]);
  }
}

// ---------------------------------------------------------------------------
// Fused window attention: one 256-thread block (8 waves) per window.
// Dynamic LDS layout (bytes), regions overlapped across barrier-separated
// stages:
//   XS  @      0 (16K)  x window bf16          [stage1-2]
//   RS  @  16384 (16K)  ref window bf16        [stage1-2]
//   QKV @  32768 (48K)  qkv(x) bf16 [64][384]  [stage2-3]
//   RK  @  81920 (16K)  k(ref) bf16 [64][128]  [stage2-3]
//   QN  @  98304 (16K)  qn bf16 [4][64][32]
//   KN  @ 114688 (16K)  kn bf16
//   KR  @ 131072 (16K)  kn_ref bf16
//   VT  @ 147456 (16K)  v^T bf16 [4][32][64]
//   SB  @      0 (64K)  scores f32 [4][64][64] [stage4+] (over dead XS/RS/QKV)
//   PB  @  65536 (32K)  probs bf16 [4][64][64] (over dead QKV/RK)
// Total 163840 B (= 160 KB exactly -> 2 workgroups per 320 KB WGP).
// ---------------------------------------------------------------------------
__global__ __launch_bounds__(256) void attn_k(
    const float* __restrict__ x, const float* __restrict__ ref,
    const unsigned short* __restrict__ wqkv, const float* __restrict__ qkvb,
    const float* __restrict__ relb, const float* __restrict__ lgs,
    const float* __restrict__ gat, float* __restrict__ img)
{
  extern __shared__ char smem[];
  unsigned short* XS  = (unsigned short*)(smem);
  unsigned short* RS  = (unsigned short*)(smem + 16384);
  unsigned short* QKV = (unsigned short*)(smem + 32768);
  unsigned short* RK  = (unsigned short*)(smem + 81920);
  unsigned short* QN  = (unsigned short*)(smem + 98304);
  unsigned short* KN  = (unsigned short*)(smem + 114688);
  unsigned short* KR  = (unsigned short*)(smem + 131072);
  unsigned short* VT  = (unsigned short*)(smem + 147456);
  float*          SB  = (float*)(smem);
  unsigned short* PB  = (unsigned short*)(smem + 65536);

  const int tid  = threadIdx.x;
  const int widx = blockIdx.x;
  const int w    = tid >> 5;
  const int lane = tid & 31;
  const int lg   = lane >> 4;
  const int ln   = lane & 15;

  // ---- stage 1: stage x / ref windows to LDS as bf16 ----
  const float* xw = x   + (long)widx * 8192;
  const float* rw = ref + (long)widx * 8192;
  for (int idx = tid; idx < 64 * 32; idx += 256) {
    int pix = idx >> 5, cq = idx & 31;
    float4 a = *(const float4*)(xw + pix * 128 + (cq << 2));
    float4 c = *(const float4*)(rw + pix * 128 + (cq << 2));
    unsigned short* p = XS + pix * 128 + (cq << 2);
    p[0] = f2bf(a.x); p[1] = f2bf(a.y); p[2] = f2bf(a.z); p[3] = f2bf(a.w);
    unsigned short* q = RS + pix * 128 + (cq << 2);
    q[0] = f2bf(c.x); q[1] = f2bf(c.y); q[2] = f2bf(c.z); q[3] = f2bf(c.w);
  }
  __syncthreads();

  // ---- stage 2a: QKV GEMM for x: 96 tiles (M=64, N=384), uniform loop ----
  for (int u = w; u < 96; u += 8) {
    int msub = u & 3;
    int nsub = u >> 2;
    v8f acc = vzero8();
    #pragma unroll
    for (int ks = 0; ks < 4; ++ks) {
      v16bf af = load_frag(XS + (msub << 4) * 128 + (ks << 5), 128, lane);
      v16bf bf = load_frag(wqkv + (nsub << 4) * 128 + (ks << 5), 128, lane);
      acc = wmma_bf16(af, bf, acc);
    }
    int ch = (nsub << 4) + ln;
    float bv = qkvb[ch];
    #pragma unroll
    for (int v = 0; v < 8; ++v) {
      int tok = (msub << 4) + v + (lg << 3);
      QKV[tok * 384 + ch] = f2bf(acc[v] + bv);
    }
  }
  // ---- stage 2b: k(ref) GEMM: 32 tiles (M=64, N=128), uniform loop ----
  for (int u = w; u < 32; u += 8) {
    int msub = u & 3;
    int nsub = u >> 2;
    v8f acc = vzero8();
    #pragma unroll
    for (int ks = 0; ks < 4; ++ks) {
      v16bf af = load_frag(RS + (msub << 4) * 128 + (ks << 5), 128, lane);
      v16bf bf = load_frag(wqkv + (128 + (nsub << 4)) * 128 + (ks << 5), 128, lane);
      acc = wmma_bf16(af, bf, acc);
    }
    int ch = (nsub << 4) + ln;
    float bv = qkvb[128 + ch];
    #pragma unroll
    for (int v = 0; v < 8; ++v) {
      int tok = (msub << 4) + v + (lg << 3);
      RK[tok * 128 + ch] = f2bf(acc[v] + bv);
    }
  }
  __syncthreads();

  // ---- stage 3: per (token, head) L2-normalize q,k,k_ref; transpose v ----
  {
    int tok = tid >> 2, h = tid & 3;
    const int qo = tok * 384 + (h << 5);
    float s; float inv;
    s = 0.f;
    for (int i = 0; i < 32; ++i) { float f = bf2f(QKV[qo + i]); s += f * f; }
    inv = 1.f / fmaxf(sqrtf(s), 1e-12f);
    for (int i = 0; i < 32; ++i)
      QN[((h << 6) + tok) * 32 + i] = f2bf(bf2f(QKV[qo + i]) * inv);
    s = 0.f;
    for (int i = 0; i < 32; ++i) { float f = bf2f(QKV[qo + 128 + i]); s += f * f; }
    inv = 1.f / fmaxf(sqrtf(s), 1e-12f);
    for (int i = 0; i < 32; ++i)
      KN[((h << 6) + tok) * 32 + i] = f2bf(bf2f(QKV[qo + 128 + i]) * inv);
    s = 0.f;
    for (int i = 0; i < 32; ++i) { float f = bf2f(RK[tok * 128 + (h << 5) + i]); s += f * f; }
    inv = 1.f / fmaxf(sqrtf(s), 1e-12f);
    for (int i = 0; i < 32; ++i)
      KR[((h << 6) + tok) * 32 + i] = f2bf(bf2f(RK[tok * 128 + (h << 5) + i]) * inv);
    for (int i = 0; i < 32; ++i)
      VT[((h << 5) + i) * 64 + tok] = QKV[qo + 256 + i];   // v stays bf16
  }

  // ---- stages 4-6, two phases: p=0 uses k(x), p=1 uses k(ref) ----
  v8f o[2][4];
  #pragma unroll
  for (int p = 0; p < 2; ++p)
    #pragma unroll
    for (int i = 0; i < 4; ++i) o[p][i] = vzero8();

  #pragma unroll
  for (int p = 0; p < 2; ++p) {
    __syncthreads();
    const unsigned short* KK = p ? KR : KN;
    // scores: S = (qn . kn) * ls + bias   (HD=32 => single WMMA per tile)
    for (int u = w; u < 16; u += 8) {
      int h = u >> 2, rb = u & 3;
      float ls = __expf(fminf(lgs[h], 4.60517019f));
      v16bf af = load_frag(QN + ((h << 6) + (rb << 4)) * 32, 32, lane);
      #pragma unroll
      for (int j = 0; j < 4; ++j) {
        v16bf bf = load_frag(KK + ((h << 6) + (j << 4)) * 32, 32, lane);
        v8f s = wmma_bf16(af, bf, vzero8());
        #pragma unroll
        for (int v = 0; v < 8; ++v) {
          int m = (rb << 4) + v + (lg << 3);     // query token
          int n = (j << 4) + ln;                 // key token
          int ridx = ((m >> 3) - (n >> 3) + 7) * 15 + ((m & 7) - (n & 7) + 7);
          SB[((h << 6) + m) * 64 + n] = s[v] * ls + relb[(ridx << 2) + h];
        }
      }
    }
    __syncthreads();
    // softmax over keys: one thread per (head,row).  Columns are scanned
    // staggered (col = (row + jj) & 63) so simultaneous lane accesses hit
    // banks (row + jj) mod 64 -> conflict-free instead of 32-way conflicts.
    {
      int h = tid >> 6, row = tid & 63;
      float* srow = SB + ((h << 6) + row) * 64;
      float mx = -1e30f;
      for (int jj = 0; jj < 64; ++jj) mx = fmaxf(mx, srow[(row + jj) & 63]);
      float sum = 0.f;
      for (int jj = 0; jj < 64; ++jj) sum += __expf(srow[(row + jj) & 63] - mx);
      float inv = 1.f / sum;
      for (int jj = 0; jj < 64; ++jj) {
        int col = (row + jj) & 63;
        PB[((h << 6) + row) * 64 + col] = f2bf(__expf(srow[col] - mx) * inv);
      }
    }
    __syncthreads();
    // P @ V : K=64 -> two WMMA k-steps per 16x16 tile
    #pragma unroll
    for (int i = 0; i < 4; ++i) {
      int u = w + (i << 3);
      int h = u >> 3, rb = (u >> 1) & 3, nb = u & 1;
      #pragma unroll
      for (int kh = 0; kh < 2; ++kh) {
        v16bf af = load_frag(PB + ((h << 6) + (rb << 4)) * 64 + (kh << 5), 64, lane);
        v16bf bf = load_frag(VT + ((h << 5) + (nb << 4)) * 64 + (kh << 5), 64, lane);
        o[p][i] = wmma_bf16(af, bf, o[p][i]);
      }
    }
  }

  // ---- stage 7: gated blend, store in image (NHWC) layout ----
  int b = widx >> 10, rem = widx & 1023, wh = rem >> 5, ww = rem & 31;
  #pragma unroll
  for (int i = 0; i < 4; ++i) {
    int u = w + (i << 3);
    int h = u >> 3, rb = (u >> 1) & 3, nb = u & 1;
    float g = 1.f / (1.f + __expf(-gat[h]));
    #pragma unroll
    for (int v = 0; v < 8; ++v) {
      int m  = (rb << 4) + v + (lg << 3);
      int ch = (h << 5) + (nb << 4) + ln;
      int yy = (wh << 3) + (m >> 3);
      int xx = (ww << 3) + (m & 7);
      img[(((b << 8) + yy) * 256 + xx) * 128 + ch] =
          (1.f - g) * o[0][i][v] + g * o[1][i][v];
    }
  }
}

// ---------------------------------------------------------------------------
// 3x3 conv as implicit GEMM over 64-pixel row strips.
// Dynamic LDS: patch (3x66x128 bf16 halo, 50688 B) + one tap-row weight slab
// (3x128x128 bf16, 98304 B) = 148992 B -> 2 workgroups / 320 KB WGP.
// Weights are staged cooperatively per dy-row (3 slabs) so the 8 waves read
// B fragments from LDS instead of 4x-duplicated global loads.
// wave w: msub=w&3 (16-pixel block), nh2=w>>2 (64-channel half), 4 accums.
// 9 taps x 4 K-steps x 4 N-subtiles = 144 WMMAs / wave.
// ---------------------------------------------------------------------------
__global__ __launch_bounds__(256) void conv3x3_k(
    const float* __restrict__ in, const unsigned short* __restrict__ wbf,
    const float* __restrict__ bias, const float* __restrict__ residual,
    float* __restrict__ out, int relu)
{
  extern __shared__ char csm[];
  unsigned short* patch = (unsigned short*)(csm);           // 50688 B
  unsigned short* wsl   = (unsigned short*)(csm + 50688);   // 98304 B

  const int tid = threadIdx.x;
  const int bid = blockIdx.x;
  const int b   = bid >> 10;
  const int rem = bid & 1023;
  const int y   = rem >> 2;
  const int x0  = (rem & 3) << 6;

  for (int idx = tid; idx < 3 * 66 * 32; idx += 256) {
    int r   = idx / (66 * 32);
    int c2  = idx % (66 * 32);
    int col = c2 >> 5, cq = c2 & 31;
    int gy  = y + r - 1;
    int gx  = x0 + col - 1;
    float4 vv = {0.f, 0.f, 0.f, 0.f};
    if (gy >= 0 && gy < 256 && gx >= 0 && gx < 256)
      vv = *(const float4*)(in + (((b << 8) + gy) * 256 + gx) * 128 + (cq << 2));
    unsigned short* p = &patch[(r * 66 + col) * 128 + (cq << 2)];
    p[0] = f2bf(vv.x); p[1] = f2bf(vv.y); p[2] = f2bf(vv.z); p[3] = f2bf(vv.w);
  }

  const int w = tid >> 5, lane = tid & 31;
  const int msub = w & 3, nh2 = w >> 2;
  const int lg = lane >> 4, ln = lane & 15;

  v8f acc[4];
  #pragma unroll
  for (int i = 0; i < 4; ++i) acc[i] = vzero8();

  for (int tg = 0; tg < 3; ++tg) {         // tap-row groups (dy = tg)
    __syncthreads();                        // protect prior slab reads / patch
    const v8u* wsrc = (const v8u*)(wbf + tg * 3 * 16384);
    v8u* wdst = (v8u*)wsl;
    for (int idx = tid; idx < 6144; idx += 256) wdst[idx] = wsrc[idx];
    __syncthreads();
    #pragma unroll
    for (int t = 0; t < 3; ++t) {           // dx = t
      const unsigned short* arow = &patch[(tg * 66 + (msub << 4) + t) * 128];
      #pragma unroll
      for (int ks = 0; ks < 4; ++ks) {
        v16bf a = load_frag(arow + (ks << 5), 128, lane);
        #pragma unroll
        for (int ns = 0; ns < 4; ++ns) {
          const unsigned short* wb =
              wsl + (((t << 7) + (nh2 << 6) + (ns << 4)) << 7) + (ks << 5);
          v16bf bf = load_frag(wb, 128, lane);
          acc[ns] = wmma_bf16(a, bf, acc[ns]);
        }
      }
    }
  }

  #pragma unroll
  for (int ns = 0; ns < 4; ++ns) {
    int ch = (nh2 << 6) + (ns << 4) + ln;
    float bv = bias[ch];
    #pragma unroll
    for (int v = 0; v < 8; ++v) {
      int px = x0 + (msub << 4) + v + (lg << 3);
      int gi = (((b << 8) + y) * 256 + px) * 128 + ch;
      float val = acc[ns][v] + bv;
      if (relu) val = fmaxf(val, 0.f);
      if (residual) val += residual[gi];
      out[gi] = val;
    }
  }
}

// ---------------------------------------------------------------------------
// 1x1 conv (proj) from image layout; writes d_out in window layout.
// Weights staged in LDS (48 KB static total).
// ---------------------------------------------------------------------------
__global__ __launch_bounds__(256) void proj_k(
    const float* __restrict__ in, const unsigned short* __restrict__ wbf,
    const float* __restrict__ bias, float* __restrict__ out)
{
  __shared__ unsigned short xs[64 * 128];      // 16384 B
  __shared__ unsigned short wp_s[128 * 128];   // 32768 B
  const int tid = threadIdx.x;
  const int bid = blockIdx.x;
  const int b   = bid >> 10;
  const int rem = bid & 1023;
  const int y   = rem >> 2;
  const int x0  = (rem & 3) << 6;

  for (int idx = tid; idx < 2048; idx += 256)
    ((v8u*)wp_s)[idx] = ((const v8u*)wbf)[idx];

  for (int idx = tid; idx < 64 * 32; idx += 256) {
    int pix = idx >> 5, cq = idx & 31;
    float4 vv = *(const float4*)(in + (((b << 8) + y) * 256 + x0 + pix) * 128 + (cq << 2));
    unsigned short* p = &xs[pix * 128 + (cq << 2)];
    p[0] = f2bf(vv.x); p[1] = f2bf(vv.y); p[2] = f2bf(vv.z); p[3] = f2bf(vv.w);
  }
  __syncthreads();

  const int w = tid >> 5, lane = tid & 31;
  const int msub = w & 3, nh2 = w >> 2;
  const int lg = lane >> 4, ln = lane & 15;

  v8f acc[4];
  #pragma unroll
  for (int i = 0; i < 4; ++i) acc[i] = vzero8();

  #pragma unroll
  for (int ks = 0; ks < 4; ++ks) {
    v16bf a = load_frag(&xs[(msub << 4) * 128 + (ks << 5)], 128, lane);
    #pragma unroll
    for (int ns = 0; ns < 4; ++ns) {
      v16bf bf = load_frag(wp_s + (((nh2 << 6) + (ns << 4)) << 7) + (ks << 5), 128, lane);
      acc[ns] = wmma_bf16(a, bf, acc[ns]);
    }
  }

  #pragma unroll
  for (int ns = 0; ns < 4; ++ns) {
    int ch = (nh2 << 6) + (ns << 4) + ln;
    float bv = bias[ch];
    #pragma unroll
    for (int v = 0; v < 8; ++v) {
      int px  = x0 + (msub << 4) + v + (lg << 3);
      int wh  = y >> 3, ihh = y & 7, ww = px >> 3, iww = px & 7;
      int wi  = (b << 10) + (wh << 5) + ww;
      int n   = (ihh << 3) + iww;
      out[(wi * 64 + n) * 128 + ch] = acc[ns][v] + bv;
    }
  }
}

// ---------------------------------------------------------------------------
// Host launcher. Workspace layout (bytes):
//   0        : wqkv bf16   (98304)
//   98304    : wproj bf16  (32768)
//   131072   : wt11 bf16   (294912)   425984: wt12   720896: wt21   1015808: wt22
//   1310720  : img0 f32 (64MB)   +64MB: img1   +128MB: img2
// Total ~194 MB.
// ---------------------------------------------------------------------------
extern "C" void kernel_launch(void* const* d_in, const int* in_sizes, int n_in,
                              void* d_out, int out_size, void* d_ws, size_t ws_size,
                              hipStream_t stream) {
  const float* x     = (const float*)d_in[0];
  const float* ref   = (const float*)d_in[1];
  const float* qkvw  = (const float*)d_in[2];
  const float* qkvb  = (const float*)d_in[3];
  const float* projw = (const float*)d_in[4];
  const float* projb = (const float*)d_in[5];
  const float* w11   = (const float*)d_in[6];
  const float* b11   = (const float*)d_in[7];
  const float* w12   = (const float*)d_in[8];
  const float* b12   = (const float*)d_in[9];
  const float* w21   = (const float*)d_in[10];
  const float* b21   = (const float*)d_in[11];
  const float* w22   = (const float*)d_in[12];
  const float* b22   = (const float*)d_in[13];
  const float* relb  = (const float*)d_in[14];
  const float* lgs   = (const float*)d_in[15];
  const float* gat   = (const float*)d_in[16];

  char* ws = (char*)d_ws;
  unsigned short* wq  = (unsigned short*)(ws + 0);
  unsigned short* wp  = (unsigned short*)(ws + 98304);
  unsigned short* c11 = (unsigned short*)(ws + 131072);
  unsigned short* c12 = (unsigned short*)(ws + 425984);
  unsigned short* c21 = (unsigned short*)(ws + 720896);
  unsigned short* c22 = (unsigned short*)(ws + 1015808);
  float* img0 = (float*)(ws + 1310720);
  float* img1 = img0 + IMGELEMS;
  float* img2 = img1 + IMGELEMS;
  float* out  = (float*)d_out;

  hipFuncSetAttribute((const void*)attn_k,
                      hipFuncAttributeMaxDynamicSharedMemorySize, 163840);
  hipFuncSetAttribute((const void*)conv3x3_k,
                      hipFuncAttributeMaxDynamicSharedMemorySize, 148992);

  prep_weights<<<576, 256, 0, stream>>>(qkvw, projw, w11, w12, w21, w22,
                                        wq, wp, c11, c12, c21, c22);
  attn_k<<<NWIN, 256, 163840, stream>>>(x, ref, wq, qkvb, relb, lgs, gat, img0);
  conv3x3_k<<<NWIN, 256, 148992, stream>>>(img0, c11, b11, nullptr, img1, 1);
  conv3x3_k<<<NWIN, 256, 148992, stream>>>(img1, c12, b12, img0,    img2, 0);
  conv3x3_k<<<NWIN, 256, 148992, stream>>>(img2, c21, b21, nullptr, img1, 1);
  conv3x3_k<<<NWIN, 256, 148992, stream>>>(img1, c22, b22, img2,    img0, 0);
  proj_k<<<NWIN, 256, 0, stream>>>(img0, wp, projb, out);
}
